// LSTMModel_23098334118250
// MI455X (gfx1250) — matrix operand order repaired and checked by reading, not compiled, not run on hardware
//
#include <hip/hip_runtime.h>

typedef __attribute__((ext_vector_type(16))) _Float16 v16h;
typedef __attribute__((ext_vector_type(8)))  float    v8f;
typedef __attribute__((ext_vector_type(2)))  __fp16   fp16x2;

#define HID   20
#define SEQ   512
#define BATCH 4096
#define SWAPX16 0x401F   // ds_swizzle group-of-32: xor=0x10, or=0, and=0x1f

// ---- fast transcendentals (gfx1250 has v_tanh_f32; guard with has_builtin) ----
__device__ __forceinline__ float tanh_fast(float x) {
#if __has_builtin(__builtin_amdgcn_tanhf)
  return __builtin_amdgcn_tanhf(x);
#else
  float e = __builtin_amdgcn_exp2f(-2.885390082f * x);       // exp(-2x)
  return (1.0f - e) * __builtin_amdgcn_rcpf(1.0f + e);
#endif
}
__device__ __forceinline__ float sigmoid_fast(float x) {
#if __has_builtin(__builtin_amdgcn_tanhf)
  return __builtin_fmaf(0.5f, __builtin_amdgcn_tanhf(0.5f * x), 0.5f);
#else
  float e = __builtin_amdgcn_exp2f(-1.442695041f * x);       // exp(-x)
  return __builtin_amdgcn_rcpf(1.0f + e);
#endif
}

__device__ __forceinline__ unsigned packf2(float a, float b) {
  fp16x2 p = __builtin_amdgcn_cvt_pkrtz(a, b);
  return __builtin_bit_cast(unsigned, p);
}

// Row permutation: padded row p (0..79) -> original gate row r = gate*20 + j.
// Lower-half D rows (w<8) carry j=0..9, upper-half rows (w>=8) carry j=10..19,
// with (i,f,g,o) of one j occupying 4 consecutive D VGPRs.
__device__ __forceinline__ int perm_row(int p) {
  int t = p >> 4, w = p & 15;
  int s, base;
  if (w < 8) { s = t * 8 + w;       base = 0;  }
  else       { s = t * 8 + (w - 8); base = 10; }
  int gate = s & 3;
  int j    = base + (s >> 2);
  return gate * HID + j;
}

__global__ void __launch_bounds__(32)
lstm_wmma_kernel(const float* __restrict__ x,
                 const float* __restrict__ W_ih, const float* __restrict__ W_hh,
                 const float* __restrict__ b_ih, const float* __restrict__ b_hh,
                 const float* __restrict__ W_lin, const float* __restrict__ b_lin,
                 float* __restrict__ out)
{
  const int  lane  = threadIdx.x;      // 0..31 (wave32)
  const int  col   = lane & 15;        // batch column within tile
  const bool upper = lane >= 16;
  const int  b0    = blockIdx.x * 16;

  // ---------------- A operand: W_ext, 5 tiles of 16x32 f16, resident in VGPRs ----
  // f16 A k order used here (contiguous, matched by B): lane holds row M = lane%16; lanes<16 carry K{0..15},
  // lanes>=16 carry K{16..31}; v16h element e -> K = e + (upper?16:0).
  v16h A[5];
#pragma unroll
  for (int t = 0; t < 5; ++t) {
    int r = perm_row(t * 16 + col);
#pragma unroll
    for (int e = 0; e < 16; ++e) {
      int k = e + (upper ? 16 : 0);
      float v;
      if (k < HID)            v = W_hh[r * HID + k];
      else if (k == HID)      v = W_ih[r];               // x column
      else if (k == HID + 1)  v = b_ih[r] + b_hh[r];     // bias column (times 1.0)
      else                    v = 0.0f;
      A[t][e] = (_Float16)v;
    }
  }

  // ---------------- per-lane constants ----------------
  const int jb = upper ? 10 : 0;       // this lane owns hidden indices jb..jb+9
  float wl[10];
#pragma unroll
  for (int q = 0; q < 10; ++q) wl[q] = W_lin[jb + q];
  const float blin = b_lin[0];

  // ---------------- state ----------------
  float    c[10];
  unsigned P[5];                       // packed f16 pairs of this lane's h values
#pragma unroll
  for (int q = 0; q < 10; ++q) c[q] = 0.0f;
#pragma unroll
  for (int t = 0; t < 5; ++t) P[t] = 0u;

  const float* xrow = x   + (size_t)(b0 + col) * SEQ;
  float*       orow = out + (size_t)(b0 + col) * SEQ;
  float xv = xrow[0];

  const v8f zc = {};                   // zero C accumulator (hoisted)

  for (int s = 0; s < SEQ; ++s) {
    float xnext = xrow[(s < SEQ - 1) ? (s + 1) : s];   // prefetch next step's x

    // upper -> lower exchange of h[j=10..15] (packed f16 pairs)
    unsigned sw0 = __builtin_amdgcn_ds_swizzle((int)P[0], SWAPX16);
    unsigned sw1 = __builtin_amdgcn_ds_swizzle((int)P[1], SWAPX16);
    unsigned sw2 = __builtin_amdgcn_ds_swizzle((int)P[2], SWAPX16);
    unsigned xp  = packf2(xv, 1.0f);                   // [x, 1.0] for K=20,21

    // B operand: h_ext column per batch lane. Lanes<16: K=0..15, lanes>=16: K=16..31.
    union { v16h v; unsigned u[8]; } B;
    B.u[0] = upper ? P[3] : P[0];
    B.u[1] = upper ? P[4] : P[1];
    B.u[2] = upper ? xp   : P[2];
    B.u[3] = upper ? 0u   : P[3];
    B.u[4] = upper ? 0u   : P[4];
    B.u[5] = upper ? 0u   : sw0;
    B.u[6] = upper ? 0u   : sw1;
    B.u[7] = upper ? 0u   : sw2;

    // gates.T [80(perm) x 16batch] = W_ext * h_ext : 5 WMMAs, C = 0
    v8f D[5];
#pragma unroll
    for (int t = 0; t < 5; ++t) {
      D[t] = __builtin_amdgcn_wmma_f32_16x16x32_f16(
          false, A[t], false, B.v, (short)0, zc, false, false);
    }

    // lane-local LSTM cell update for this lane's 10 hidden units
    float hq[10];
    float outacc = 0.0f;
#pragma unroll
    for (int q = 0; q < 10; ++q) {
      const int tq = q >> 1, off = (q & 1) * 4;
      float gi = D[tq][off + 0];
      float gf = D[tq][off + 1];
      float gg = D[tq][off + 2];
      float go = D[tq][off + 3];
      float cs = __builtin_fmaf(sigmoid_fast(gf), c[q],
                                sigmoid_fast(gi) * tanh_fast(gg));
      c[q]  = cs;
      float h = sigmoid_fast(go) * tanh_fast(cs);
      hq[q] = h;
      outacc = __builtin_fmaf(wl[q], h, outacc);
    }
#pragma unroll
    for (int t = 0; t < 5; ++t) P[t] = packf2(hq[2 * t], hq[2 * t + 1]);

    // output: full dot = own 10 terms + partner's 10 terms (both halves store same value)
    int   swo = __builtin_amdgcn_ds_swizzle(__builtin_bit_cast(int, outacc), SWAPX16);
    float tot = outacc + __builtin_bit_cast(float, swo);
    orow[s] = tot + blin;

    xv = xnext;
  }
}

extern "C" void kernel_launch(void* const* d_in, const int* in_sizes, int n_in,
                              void* d_out, int out_size, void* d_ws, size_t ws_size,
                              hipStream_t stream) {
  (void)in_sizes; (void)n_in; (void)d_ws; (void)ws_size; (void)out_size;
  const float* x     = (const float*)d_in[0];
  const float* W_ih  = (const float*)d_in[1];
  const float* W_hh  = (const float*)d_in[2];
  const float* b_ih  = (const float*)d_in[3];
  const float* b_hh  = (const float*)d_in[4];
  const float* W_lin = (const float*)d_in[5];
  const float* b_lin = (const float*)d_in[6];
  float* out = (float*)d_out;

  dim3 grid(BATCH / 16);   // 256 waves, one 16-row batch tile each
  dim3 block(32);          // one wave32 per workgroup
  lstm_wmma_kernel<<<grid, block, 0, stream>>>(x, W_ih, W_hh, b_ih, b_hh,
                                               W_lin, b_lin, out);
}
